// Head_3092376453739
// MI455X (gfx1250) — matrix operand order repaired and checked by reading, not compiled, not run on hardware
//
#include <hip/hip_runtime.h>
#include <math.h>

// MI455X / gfx1250 fused causal attention (B=4, T=4096, C=1024, H=64)
// Phase 0: pre-swizzle W into WMMA B-fragment layout (f16).
// Phase 1: QKV projection (f32 -> f16), 2x-unrolled ping-pong pipeline.
// Phase 2: flash-attention, online f32 softmax, ping-pong K prefetch.

typedef __attribute__((ext_vector_type(16))) _Float16 v16h;
typedef __attribute__((ext_vector_type(8)))  _Float16 v8h;
typedef __attribute__((ext_vector_type(8)))  float    v8f;
typedef __attribute__((ext_vector_type(4)))  float    v4f;

#define BATCH   4
#define SEQ     4096
#define CDIM    1024
#define HDIM    64
#define TOKENS  (BATCH * SEQ)     // 16384
#define KCHUNKS (CDIM / 32)       // 32 K-chunks per projection
#define FRAG_HALVES (32 * 16)     // one fragment: 32 lanes x 16 halves
#define WFRAG_PER_MAT (KCHUNKS * 4)

__device__ __forceinline__ v8f wmma_f16(v16h a, v16h b, v8f c) {
  // (neg_a, A, neg_b, B, c_mod, C, reuse_a, reuse_b)
  return __builtin_amdgcn_wmma_f32_16x16x32_f16(false, a, false, b, (short)0, c, false, false);
}

// ---------------------------------------------------------------------------
// Kernel 0: pre-swizzle Wq/Wk/Wv (f32 [C][H]) into B-fragment layout (f16).
// Fragment (mat, kc, n): lane l, elem i = W[(kc*32 + (i>>3)*16 + (l>>4)*8 + (i&7))*64 + n*16 + (l&15)]
// ---------------------------------------------------------------------------
__global__ __launch_bounds__(256) void wfrag_prep_kernel(
    const float* __restrict__ Wq, const float* __restrict__ Wk,
    const float* __restrict__ Wv, _Float16* __restrict__ wfrag)
{
  const int tid  = threadIdx.x;
  const int lane = tid & 31;
  const int warp = tid >> 5;
  const int frag = blockIdx.x * 8 + warp;         // 0..383
  const int mat  = frag / WFRAG_PER_MAT;
  const int rem  = frag % WFRAG_PER_MAT;
  const int kc   = rem >> 2;
  const int n    = rem & 3;
  const float* W = (mat == 0) ? Wq : (mat == 1 ? Wk : Wv);

  const int row = lane & 15;
  const int grp = lane >> 4;

  v16h f;
#pragma unroll
  for (int i = 0; i < 16; ++i)
    f[i] = (_Float16)W[(size_t)(kc * 32 + (i < 8 ? grp * 8 + i : 16 + grp * 8 + (i - 8))) * HDIM + n * 16 + row];

  *(v16h*)(wfrag + ((size_t)frag * 32 + lane) * 16) = f;
}

// ---------------------------------------------------------------------------
// Kernel 1: q = x@Wq, k = x@Wk, v = x@Wv  (outputs f16; v transposed)
// 2x-unrolled ping-pong: no buffer-rotation register copies.
// ---------------------------------------------------------------------------
__device__ __forceinline__ void proj_load(const float* xrow, const _Float16* wf,
                                          int kc, int kb8, v4f* xc, v16h* bc) {
  const int k0 = kc * 32;
  xc[0] = *(const v4f*)(xrow + k0 + kb8);
  xc[1] = *(const v4f*)(xrow + k0 + kb8 + 4);
  xc[2] = *(const v4f*)(xrow + k0 + kb8 + 16);
  xc[3] = *(const v4f*)(xrow + k0 + kb8 + 20);
#pragma unroll
  for (int n = 0; n < 4; ++n)
    bc[n] = *(const v16h*)(wf + ((size_t)kc * 4 + n) * FRAG_HALVES);
}

__device__ __forceinline__ void proj_compute(const v4f* xc, const v16h* bc, v8f* acc) {
  v16h a;  // A elems 0..7 -> K=kb8+0..7, 8..15 -> K=kb8+16..23
#pragma unroll
  for (int i = 0; i < 4; ++i) {
    a[i]      = (_Float16)xc[0][i];
    a[4 + i]  = (_Float16)xc[1][i];
    a[8 + i]  = (_Float16)xc[2][i];
    a[12 + i] = (_Float16)xc[3][i];
  }
#pragma unroll
  for (int n = 0; n < 4; ++n)
    acc[n] = wmma_f16(a, bc[n], acc[n]);
}

__global__ __launch_bounds__(256) void qkv_proj_kernel(
    const float* __restrict__ x, const _Float16* __restrict__ wfrag,
    _Float16* __restrict__ q, _Float16* __restrict__ k,
    _Float16* __restrict__ vT)
{
  const int tid  = threadIdx.x;
  const int lane = tid & 31;
  const int warp = tid >> 5;
  const int mat  = blockIdx.y;                    // 0=q 1=k 2=v

  const int tile  = blockIdx.x * 8 + warp;        // 0..1023
  const int tbase = tile * 16;

  const int row = lane & 15;
  const int grp = lane >> 4;
  const int kb8 = grp * 8;

  const float*    xrow = x + (size_t)(tbase + row) * CDIM;
  const _Float16* wf   = wfrag + (size_t)mat * WFRAG_PER_MAT * FRAG_HALVES
                               + (size_t)lane * 16;

  v8f acc[4] = {v8f{}, v8f{}, v8f{}, v8f{}};

  v4f  xA[4], xB[4];
  v16h bA[4], bB[4];
  proj_load(xrow, wf, 0, kb8, xA, bA);

  for (int kc = 0; kc < KCHUNKS; kc += 2) {
    proj_load(xrow, wf, kc + 1, kb8, xB, bB);               // prefetch odd chunk
    proj_compute(xA, bA, acc);                              // consume even chunk
    proj_load(xrow, wf, (kc + 2) & (KCHUNKS - 1), kb8, xA, bA); // prefetch next even
    proj_compute(xB, bB, acc);                              // consume odd chunk
  }

  if (mat < 2) {
    _Float16* out = (mat == 0) ? q : k;           // row-major [TOKENS][64]
#pragma unroll
    for (int n = 0; n < 4; ++n)
#pragma unroll
      for (int r = 0; r < 8; ++r) {
        const int t = tbase + r + 8 * grp;        // C layout: M = r + 8*grp
        out[(size_t)t * HDIM + n * 16 + row] = (_Float16)acc[n][r];
      }
  } else {
    const int b  = tbase / SEQ;                   // tiles never cross batch
    const int tl = tbase % SEQ;
#pragma unroll
    for (int n = 0; n < 4; ++n)
#pragma unroll
      for (int r = 0; r < 8; ++r) {
        const int col = n * 16 + row;
        vT[((size_t)b * HDIM + col) * SEQ + tl + r + 8 * grp] = (_Float16)acc[n][r];
      }
  }
}

// ---------------------------------------------------------------------------
// Kernel 2: flash attention. One wave per 16-row query tile (1024 tiles).
// 2x-unrolled ping-pong K prefetch; V loads at iteration top are hidden by
// the softmax VALU section.
// ---------------------------------------------------------------------------
__global__ __launch_bounds__(256) void flash_attn_kernel(
    const _Float16* __restrict__ q, const _Float16* __restrict__ k,
    const _Float16* __restrict__ vT, float* __restrict__ out)
{
  __shared__ _Float16 lds_p[8][16 * 32];  // per-wave P transpose buffer

  const int tid  = threadIdx.x;
  const int lane = tid & 31;
  const int warp = tid >> 5;
  const int wg   = blockIdx.x * 8 + warp; // 0..1023
  const int b    = wg >> 8;
  const int qt   = wg & 255;
  const int tbase = wg * 16;

  const int row = lane & 15;
  const int grp = lane >> 4;
  const int kb8 = grp * 8;
  const float scale = 1.0f / 32.0f;       // 1/sqrt(C)

  // ---- Q A-fragments (16x64 -> two 16x32 frags) ----
  const _Float16* qrow = q + (size_t)(tbase + row) * HDIM;
  v16h aq0, aq1;
  {
    v8h l0 = *(const v8h*)(qrow + grp * 16);
    v8h h0 = *(const v8h*)(qrow + grp * 16 + 8);
    v8h l1 = *(const v8h*)(qrow + 32 + grp * 16);
    v8h h1 = *(const v8h*)(qrow + 32 + grp * 16 + 8);
#pragma unroll
    for (int i = 0; i < 8; ++i) {
      aq0[i] = l0[i]; aq0[8 + i] = h0[i];
      aq1[i] = l1[i]; aq1[8 + i] = h1[i];
    }
  }

  v8f o0 = {}, o1 = {}, o2 = {}, o3 = {};
  float m_r[8], l_r[8];
#pragma unroll
  for (int r = 0; r < 8; ++r) { m_r[r] = -INFINITY; l_r[r] = 0.0f; }

  const _Float16* kbat = k  + (size_t)b * SEQ * HDIM;
  const _Float16* vbat = vT + (size_t)b * HDIM * SEQ;

  const int nblk = (qt >> 1) + 1;         // 32-key blocks up to the diagonal

  auto load_kfrags = [&](int key0, v16h& k0a, v16h& k0b, v16h& k1a, v16h& k1b) {
    const _Float16* kr0 = kbat + (size_t)(key0 + row) * HDIM;
    const _Float16* kr1 = kbat + (size_t)(key0 + 16 + row) * HDIM;
    k0a = *(const v16h*)(kr0 + grp * 16);        // d = 0..31
    k0b = *(const v16h*)(kr0 + 32 + grp * 16);   // d = 32..63
    k1a = *(const v16h*)(kr1 + grp * 16);
    k1b = *(const v16h*)(kr1 + 32 + grp * 16);
  };

  // one attention block: consume K frags (ck*), prefetch block keyn into nk*
  auto attn_block = [&](int key0, int keyn,
                        v16h& ck0a, v16h& ck0b, v16h& ck1a, v16h& ck1b,
                        v16h& nk0a, v16h& nk0b, v16h& nk1a, v16h& nk1b) {
    // ---- V loads for current block (latency hidden by softmax) ----
    const _Float16* vb = vbat + key0 + grp * 16;
    v16h bv0 = *(const v16h*)(vb + (size_t)(0  + row) * SEQ);
    v16h bv1 = *(const v16h*)(vb + (size_t)(16 + row) * SEQ);
    v16h bv2 = *(const v16h*)(vb + (size_t)(32 + row) * SEQ);
    v16h bv3 = *(const v16h*)(vb + (size_t)(48 + row) * SEQ);

    // ---- S = Q @ K^T for 32 keys ----
    v8f s0 = {}, s1 = {};
    s0 = wmma_f16(aq0, ck0a, s0);
    s0 = wmma_f16(aq1, ck0b, s0);
    s1 = wmma_f16(aq0, ck1a, s1);
    s1 = wmma_f16(aq1, ck1b, s1);

    // ---- prefetch next block's K fragments ----
    load_kfrags(keyn, nk0a, nk0b, nk1a, nk1b);

    // ---- scale, causal mask, online softmax ----
#pragma unroll
    for (int r = 0; r < 8; ++r) {
      const int qr  = qt * 16 + r + 8 * grp;
      const int kc0 = key0 + row;
      const int kc1 = key0 + 16 + row;
      float a0 = s0[r] * scale; if (kc0 > qr) a0 = -INFINITY;
      float a1 = s1[r] * scale; if (kc1 > qr) a1 = -INFINITY;

      float m = fmaxf(a0, a1);
      m = fmaxf(m, __shfl_xor(m, 1, 16));
      m = fmaxf(m, __shfl_xor(m, 2, 16));
      m = fmaxf(m, __shfl_xor(m, 4, 16));
      m = fmaxf(m, __shfl_xor(m, 8, 16));
      const float mn    = fmaxf(m_r[r], m);
      const float alpha = __expf(m_r[r] - mn);  // 0 on first block
      m_r[r] = mn;

      const float p0 = __expf(a0 - mn);
      const float p1 = __expf(a1 - mn);
      float rs = p0 + p1;
      rs += __shfl_xor(rs, 1, 16);
      rs += __shfl_xor(rs, 2, 16);
      rs += __shfl_xor(rs, 4, 16);
      rs += __shfl_xor(rs, 8, 16);
      l_r[r] = l_r[r] * alpha + rs;

      lds_p[warp][(r + 8 * grp) * 32 + row]      = (_Float16)p0;
      lds_p[warp][(r + 8 * grp) * 32 + 16 + row] = (_Float16)p1;

      o0[r] *= alpha; o1[r] *= alpha; o2[r] *= alpha; o3[r] *= alpha;
    }

    // same-wave LDS RAW: wait for DS writes before fragment reads
    asm volatile("s_wait_dscnt 0" ::: "memory");

    // ---- P back in A-matrix 16x32 layout ----
    v16h ap;
    {
      const _Float16* pr = &lds_p[warp][row * 32];
      v8h plo = *(const v8h*)(pr + grp * 16);
      v8h phi = *(const v8h*)(pr + grp * 16 + 8);
#pragma unroll
      for (int i = 0; i < 8; ++i) { ap[i] = plo[i]; ap[8 + i] = phi[i]; }
    }

    // ---- O += P @ V ----
    o0 = wmma_f16(ap, bv0, o0);
    o1 = wmma_f16(ap, bv1, o1);
    o2 = wmma_f16(ap, bv2, o2);
    o3 = wmma_f16(ap, bv3, o3);
  };

  // ---- ping-pong over 32-key blocks ----
  v16h Ak0a, Ak0b, Ak1a, Ak1b;   // buffer set A
  v16h Bk0a, Bk0b, Bk1a, Bk1b;   // buffer set B
  load_kfrags(0, Ak0a, Ak0b, Ak1a, Ak1b);

  int j = 0;
  for (; j + 1 < nblk; j += 2) {
    attn_block(j * 32, (j + 1) * 32,
               Ak0a, Ak0b, Ak1a, Ak1b, Bk0a, Bk0b, Bk1a, Bk1b);
    const int keyn2 = ((j + 2 < nblk) ? j + 2 : j + 1) * 32;
    attn_block((j + 1) * 32, keyn2,
               Bk0a, Bk0b, Bk1a, Bk1b, Ak0a, Ak0b, Ak1a, Ak1b);
  }
  if (j < nblk) {
    // tail: prefetch target regs are dead afterwards (self-prefetch)
    attn_block(j * 32, j * 32,
               Ak0a, Ak0b, Ak1a, Ak1b, Ak0a, Ak0b, Ak1a, Ak1b);
  }

  // ---- normalize and store ----
#pragma unroll
  for (int r = 0; r < 8; ++r) {
    const float inv = 1.0f / l_r[r];
    const size_t t = (size_t)(tbase + r + 8 * grp);
    out[t * HDIM + 0  + row] = o0[r] * inv;
    out[t * HDIM + 16 + row] = o1[r] * inv;
    out[t * HDIM + 32 + row] = o2[r] * inv;
    out[t * HDIM + 48 + row] = o3[r] * inv;
  }
}

// ---------------------------------------------------------------------------
extern "C" void kernel_launch(void* const* d_in, const int* in_sizes, int n_in,
                              void* d_out, int out_size, void* d_ws, size_t ws_size,
                              hipStream_t stream) {
  const float* x  = (const float*)d_in[0];   // [B,T,C]
  const float* Wq = (const float*)d_in[1];   // [C,H]
  const float* Wk = (const float*)d_in[2];
  const float* Wv = (const float*)d_in[3];
  float* out = (float*)d_out;                // [B,T,H] f32

  // workspace: q, k (row-major f16), vT (transposed f16), W fragments
  _Float16* q  = (_Float16*)d_ws;
  _Float16* kk = q  + (size_t)TOKENS * HDIM;
  _Float16* vT = kk + (size_t)TOKENS * HDIM;
  _Float16* wf = vT + (size_t)TOKENS * HDIM;  // 3*128KB of fragments

  wfrag_prep_kernel<<<(3 * WFRAG_PER_MAT) / 8, 256, 0, stream>>>(Wq, Wk, Wv, wf);

  dim3 g1(TOKENS / 16 / 8, 3, 1);             // 128 x 3 blocks, 8 waves each
  qkv_proj_kernel<<<g1, 256, 0, stream>>>(x, wf, q, kk, vT);

  flash_attn_kernel<<<TOKENS / 16 / 8, 256, 0, stream>>>(q, kk, vT, out);
}
